// ARSLMModel_80685255623091
// MI455X (gfx1250) — compile-verified, matrix operand order checked by reading, not run on hardware
//
#include <hip/hip_runtime.h>
#include <hip/hip_bf16.h>

// ---------------------------------------------------------------------------
// Problem constants (from reference): V=32000 E=256 H=256 L=2 B=2 T=2048
// ---------------------------------------------------------------------------
#define VV    32000
#define EE    256
#define HH    256
#define CTXW  768          // 2H + E
#define LAYERS 2
#define TT    2048
#define BB    2
#define NTILES (VV / 16)   // 2000 n-tiles of 16
#define MTILES ((BB * TT) / 16) // 256 m-tiles of 16
#define KBLKS  (HH / 32)   // 8 k-steps of 32 (bf16 WMMA)

typedef __attribute__((ext_vector_type(16))) __bf16 v16bf;
typedef __attribute__((ext_vector_type(8)))  float  v8f;

// ---------------------------------------------------------------------------
// K1: embedding gather  x[b,t,:] = emb[ids[b,t],:]
// ---------------------------------------------------------------------------
__global__ void embed_kernel(const int* __restrict__ ids,
                             const float* __restrict__ emb,
                             float* __restrict__ x) {
  const int bt = blockIdx.x;            // 0 .. B*T-1
  const int e  = threadIdx.x;           // 0 .. 255
  x[(size_t)bt * EE + e] = emb[(size_t)ids[bt] * EE + e];
}

// ---------------------------------------------------------------------------
// K2: convert head_w f32 [256,32000] -> bf16, swizzled into the CDNA5 WMMA
// 16-bit B-operand per-lane layout: packed element index
//   ((ntile*8 + kblk)*32 + lane)*16 + half
// where lane = n + 16*k[3],  half = k[2:0] + 8*k[4]  (k within 32-block).
// ---------------------------------------------------------------------------
__global__ void pack_headw_kernel(const float* __restrict__ w,
                                  __hip_bfloat16* __restrict__ Bp) {
  size_t idx = (size_t)blockIdx.x * 256 + threadIdx.x; // one packed bf16 each
  int hh = (int)(idx & 15);
  size_t r = idx >> 4;
  int lane = (int)(r & 31); r >>= 5;
  int kb   = (int)(r & 7);
  size_t nblk = r >> 3;
  int n = lane & 15;
  int k = kb * 32 + ((lane >> 4) & 1) * 8 + ((hh >> 3) << 4) + (hh & 7);
  Bp[idx] = __float2bfloat16(w[(size_t)k * VV + nblk * 16 + n]);
}

// ---------------------------------------------------------------------------
// K3: serial recurrent scan. One block per batch, 1024 threads (32 waves,
// wave32). j = tid&255 output column, s = tid>>8 K-slice (4-way K split).
// ---------------------------------------------------------------------------
__global__ __launch_bounds__(1024) void rnn_scan_kernel(
    const float* __restrict__ x,
    const float* __restrict__ w1, const float* __restrict__ b1,
    const float* __restrict__ w2, const float* __restrict__ b2,
    const float* __restrict__ gw, const float* __restrict__ gb,
    const float* __restrict__ lng, const float* __restrict__ lnb,
    float* __restrict__ hs) {
  const int b   = blockIdx.x;
  const int tid = threadIdx.x;
  const int j   = tid & (HH - 1);
  const int s   = tid >> 8;
  const int ln  = tid & 31;
  const int wv  = tid >> 5;

  __shared__ float h1s[LAYERS][HH], h2s[LAYERS][HH];
  __shared__ float ctx[CTXW];
  __shared__ float inp[HH];
  __shared__ float psum[4][HH];
  __shared__ float c1[HH];
  __shared__ float redbuf[32];
  __shared__ float g_sh, mean_sh, var_sh;

  if (s == 0)
    for (int l = 0; l < LAYERS; ++l) { h1s[l][j] = 0.f; h2s[l][j] = 0.f; }
  __syncthreads();

  for (int t = 0; t < TT; ++t) {
    if (s == 0) inp[j] = x[((size_t)b * TT + t) * EE + j];
    __syncthreads();

    for (int l = 0; l < LAYERS; ++l) {
      // ctx = [h1 | h2 | inp]
      if (s == 0) {
        ctx[j]          = h1s[l][j];
        ctx[HH + j]     = h2s[l][j];
        ctx[2 * HH + j] = inp[j];
      }
      __syncthreads();

      // cand layer 1: [768] dot, 4-way K-sliced
      const float* W1l = w1 + (size_t)l * CTXW * HH;
      const float* gwl = gw + l * CTXW;
      float acc = 0.f;
      const int k0 = s * (CTXW / 4);
      #pragma unroll 8
      for (int k = 0; k < CTXW / 4; ++k)
        acc = fmaf(ctx[k0 + k], W1l[(size_t)(k0 + k) * HH + j], acc);
      psum[s][j] = acc;

      // gate dot over all 1024 threads
      float gp = (tid < CTXW) ? ctx[tid] * gwl[tid] : 0.f;
      #pragma unroll
      for (int off = 16; off; off >>= 1) gp += __shfl_xor(gp, off, 32);
      if (ln == 0) redbuf[wv] = gp;
      __syncthreads();

      if (tid == 0) {
        float sum = 0.f;
        for (int w = 0; w < 32; ++w) sum += redbuf[w];
        g_sh = 1.f / (1.f + __expf(-(sum + gb[l])));
      }
      if (s == 0) {
        float v = psum[0][j] + psum[1][j] + psum[2][j] + psum[3][j] + b1[l * HH + j];
        c1[j] = fmaxf(v, 0.f);
      }
      __syncthreads();

      // cand layer 2: [256] dot, 4-way K-sliced
      const float* W2l = w2 + (size_t)l * HH * HH;
      float acc2 = 0.f;
      const int q0 = s * (HH / 4);
      #pragma unroll 8
      for (int k = 0; k < HH / 4; ++k)
        acc2 = fmaf(c1[q0 + k], W2l[(size_t)(q0 + k) * HH + j], acc2);
      psum[s][j] = acc2;
      __syncthreads();

      float pre = 0.f;
      if (s == 0) {
        float cand = psum[0][j] + psum[1][j] + psum[2][j] + psum[3][j] + b2[l * HH + j];
        pre = h1s[l][j] + g_sh * cand + 0.1f * inp[j];
      }
      // layernorm mean (waves 0..7 hold the 256 values)
      float mv = (s == 0) ? pre : 0.f;
      #pragma unroll
      for (int off = 16; off; off >>= 1) mv += __shfl_xor(mv, off, 32);
      if (s == 0 && ln == 0) redbuf[wv] = mv;
      __syncthreads();
      if (tid == 0) {
        float sum = 0.f; for (int w = 0; w < 8; ++w) sum += redbuf[w];
        mean_sh = sum * (1.f / HH);
      }
      __syncthreads();
      float dv = (s == 0) ? (pre - mean_sh) : 0.f;
      float vv = dv * dv;
      #pragma unroll
      for (int off = 16; off; off >>= 1) vv += __shfl_xor(vv, off, 32);
      if (s == 0 && ln == 0) redbuf[wv] = vv;
      __syncthreads();
      if (tid == 0) {
        float sum = 0.f; for (int w = 0; w < 8; ++w) sum += redbuf[w];
        var_sh = sum * (1.f / HH);
      }
      __syncthreads();
      if (s == 0) {
        float hval = (pre - mean_sh) * rsqrtf(var_sh + 1e-5f) * lng[l * HH + j]
                     + lnb[l * HH + j];
        h2s[l][j] = h1s[l][j];
        h1s[l][j] = hval;
        inp[j]    = hval;
        if (l == LAYERS - 1) hs[((size_t)b * TT + t) * HH + j] = hval;
      }
      __syncthreads();
    }
  }
}

// ---------------------------------------------------------------------------
// K4: attention score MLP: scores[bt] = tanh(h@W1+b1)@w2 + b2
// ---------------------------------------------------------------------------
__global__ __launch_bounds__(256) void attn_score_kernel(
    const float* __restrict__ hs,
    const float* __restrict__ w1, const float* __restrict__ b1,
    const float* __restrict__ w2, const float* __restrict__ b2,
    float* __restrict__ scores) {
  const int bt = blockIdx.x;
  const int j  = threadIdx.x;
  __shared__ float hsh[HH];
  __shared__ float red[8];
  hsh[j] = hs[(size_t)bt * HH + j];
  __syncthreads();
  float a = b1[j];
  #pragma unroll 8
  for (int k = 0; k < HH; ++k) a = fmaf(hsh[k], w1[(size_t)k * HH + j], a);
  float v = tanhf(a) * w2[j];
  #pragma unroll
  for (int off = 16; off; off >>= 1) v += __shfl_xor(v, off, 32);
  if ((j & 31) == 0) red[j >> 5] = v;
  __syncthreads();
  if (j == 0) {
    float sum = 0.f; for (int w = 0; w < 8; ++w) sum += red[w];
    scores[bt] = sum + b2[0];
  }
}

// ---------------------------------------------------------------------------
// K5: causal-prefix softmax as an online-softmax streaming scan, fused with
// attended = hs + ctx and bf16 A-operand packing into the CDNA5 WMMA layout.
// One block per batch, thread j owns hidden channel j.
// ---------------------------------------------------------------------------
__global__ __launch_bounds__(256) void attn_stream_kernel(
    const float* __restrict__ hs, const float* __restrict__ scores,
    __hip_bfloat16* __restrict__ Ap) {
  const int b = blockIdx.x;
  const int j = threadIdx.x;
  const int kb = j >> 5, kk = j & 31;
  const int hlf = (kk & 7) + ((kk >> 4) << 3);
  const int lhi = ((kk >> 3) & 1) * 16;
  float n = 0.f, d = 0.f, m = -1e30f;   // every thread tracks identical m,d
  for (int t = 0; t < TT; ++t) {
    float sc   = scores[b * TT + t];
    float mnew = fmaxf(m, sc);
    float scl  = __expf(m - mnew);
    float e    = __expf(sc - mnew);
    float hv   = hs[((size_t)b * TT + t) * HH + j];
    n = n * scl + e * hv;
    d = d * scl + e;
    m = mnew;
    float att = hv + n / d;
    int row = b * TT + t;
    int mblk = row >> 4, mr = row & 15;
    int lane = mr + lhi;
    Ap[(((size_t)mblk * KBLKS + kb) * 32 + lane) * 16 + hlf] = __float2bfloat16(att);
  }
}

// ---------------------------------------------------------------------------
// K6: head GEMM  out[4096,32000] = A(bf16) @ B(bf16) + bias,  f32 accum via
// v_wmma_f32_16x16x32_bf16. One wave per (m-tile x 4 n-tiles); 32 WMMA/wave.
// ---------------------------------------------------------------------------
__global__ __launch_bounds__(256) void head_gemm_kernel(
    const __hip_bfloat16* __restrict__ Ap,
    const __hip_bfloat16* __restrict__ Bp,
    const float* __restrict__ bias,
    float* __restrict__ out) {
  const int lane = threadIdx.x & 31;
  const int wv   = threadIdx.x >> 5;
  const int wid  = blockIdx.x * 8 + wv;          // 0 .. 127999
  const int mblk = wid / (NTILES / 4);           // 0 .. 255
  const int ngrp = wid % (NTILES / 4);           // 0 .. 499 (4 n-tiles each)

  const v16bf* A = (const v16bf*)(const void*)Ap + (size_t)mblk * KBLKS * 32 + lane;
  const v16bf* Bb = (const v16bf*)(const void*)Bp
                    + (size_t)(ngrp * 4) * KBLKS * 32 + lane;

  v8f acc0 = {}, acc1 = {}, acc2 = {}, acc3 = {};
  #pragma unroll
  for (int kb = 0; kb < KBLKS; ++kb) {
    v16bf a  = A[kb * 32];
    const v16bf* B = Bb + kb * 32;
    __builtin_prefetch((const void*)(B + 32), 0, 1);  // next K-slab of B-tile 0
    v16bf b0 = B[0 * KBLKS * 32];
    v16bf b1 = B[1 * KBLKS * 32];
    v16bf b2 = B[2 * KBLKS * 32];
    v16bf b3 = B[3 * KBLKS * 32];
    acc0 = __builtin_amdgcn_wmma_f32_16x16x32_bf16(false, a, false, b0, (short)0, acc0, false, false);
    acc1 = __builtin_amdgcn_wmma_f32_16x16x32_bf16(false, a, false, b1, (short)0, acc1, false, false);
    acc2 = __builtin_amdgcn_wmma_f32_16x16x32_bf16(false, a, false, b2, (short)0, acc2, false, false);
    acc3 = __builtin_amdgcn_wmma_f32_16x16x32_bf16(false, a, false, b3, (short)0, acc3, false, false);
  }

  // C/D layout: lane = n + 16*(m>=8), vgpr r -> m = r + 8*(lane>=16)
  const int cn = lane & 15;
  const int rh = lane >> 4;
  #pragma unroll
  for (int r = 0; r < 8; ++r) {
    int row = mblk * 16 + rh * 8 + r;
    size_t ro = (size_t)row * VV;
    int c0 = (ngrp * 4 + 0) * 16 + cn;
    int c1c = (ngrp * 4 + 1) * 16 + cn;
    int c2c = (ngrp * 4 + 2) * 16 + cn;
    int c3c = (ngrp * 4 + 3) * 16 + cn;
    out[ro + c0]  = acc0[r] + bias[c0];
    out[ro + c1c] = acc1[r] + bias[c1c];
    out[ro + c2c] = acc2[r] + bias[c2c];
    out[ro + c3c] = acc3[r] + bias[c3c];
  }
}

// ---------------------------------------------------------------------------
// Launch
// ---------------------------------------------------------------------------
extern "C" void kernel_launch(void* const* d_in, const int* in_sizes, int n_in,
                              void* d_out, int out_size, void* d_ws, size_t ws_size,
                              hipStream_t stream) {
  (void)in_sizes; (void)n_in; (void)out_size; (void)ws_size;
  const int*   ids   = (const int*)  d_in[0];
  const float* emb   = (const float*)d_in[1];
  const float* cw1   = (const float*)d_in[2];
  const float* cb1   = (const float*)d_in[3];
  const float* cw2   = (const float*)d_in[4];
  const float* cb2   = (const float*)d_in[5];
  const float* gw    = (const float*)d_in[6];
  const float* gb    = (const float*)d_in[7];
  const float* lng   = (const float*)d_in[8];
  const float* lnb   = (const float*)d_in[9];
  const float* aw1   = (const float*)d_in[10];
  const float* ab1   = (const float*)d_in[11];
  const float* aw2   = (const float*)d_in[12];
  const float* ab2   = (const float*)d_in[13];
  const float* headw = (const float*)d_in[14];
  const float* headb = (const float*)d_in[15];

  char* ws = (char*)d_ws;
  float* x      = (float*)(ws);                                  //  4 MB
  float* hs     = (float*)(ws + (4u << 20));                     //  4 MB
  float* scores = (float*)(ws + (8u << 20));                     // 16 KB
  __hip_bfloat16* Ap = (__hip_bfloat16*)(ws + (8u << 20) + (64u << 10)); // 2 MB
  __hip_bfloat16* Bp = (__hip_bfloat16*)(ws + (8u << 20) + (64u << 10) + (2u << 20)); // 16 MB

  embed_kernel<<<BB * TT, EE, 0, stream>>>(ids, emb, x);
  pack_headw_kernel<<<(HH * VV) / 256, 256, 0, stream>>>(headw, Bp);
  rnn_scan_kernel<<<BB, 1024, 0, stream>>>(x, cw1, cb1, cw2, cb2,
                                           gw, gb, lng, lnb, hs);
  attn_score_kernel<<<BB * TT, HH, 0, stream>>>(hs, aw1, ab1, aw2, ab2, scores);
  attn_stream_kernel<<<BB, HH, 0, stream>>>(hs, scores, Ap);
  head_gemm_kernel<<<(MTILES * (NTILES / 4)) / 8, 256, 0, stream>>>(Ap, Bp, headb,
                                                                   (float*)d_out);
}
